// ChebGNN_80032420594056
// MI455X (gfx1250) — matrix operand order repaired and checked by reading, not compiled, not run on hardware
//
#include <hip/hip_runtime.h>
#include <hip/hip_bf16.h>
#include <math.h>

// ---------------------------------------------------------------------------
// ChebGNN on MI455X (gfx1250, wave32).
// GEMMs: v_wmma_f32_16x16x32_bf16, f32 accumulate. Each wave owns TWO 16x16
// output tiles so two independent WMMA chains interleave and fill the
// WMMA->WMMA hazard slots. Branch-free row clamping (N % 16 == 0 here).
// Edge propagation: gather + global_atomic_add_f32 scatter (feature buffers
// are ~51MB each -> resident in the 192MB L2).
// ---------------------------------------------------------------------------

typedef __attribute__((ext_vector_type(16))) __bf16 v16bf;
typedef __attribute__((ext_vector_type(8)))  __bf16 v8bf;
typedef __attribute__((ext_vector_type(8)))  float  v8f;

#define HIDC   256
#define KCHEB  5
#define NGRAPH 64
#define FIN    128

// ---------------- elementwise helpers ----------------
__global__ void fill_zero_k(float* p, int n) {
  int i = blockIdx.x * blockDim.x + threadIdx.x;
  if (i < n) p[i] = 0.f;
}
__global__ void copy_neg_k(const float* __restrict__ a, float* __restrict__ o, int n) {
  int i = blockIdx.x * blockDim.x + threadIdx.x;
  if (i < n) o[i] = -a[i];
}
__global__ void neg_inplace_k(float* p, int n) {
  int i = blockIdx.x * blockDim.x + threadIdx.x;
  if (i < n) p[i] = -p[i];
}
__global__ void relu_k(float* p, int n) {
  int i = blockIdx.x * blockDim.x + threadIdx.x;
  if (i < n) p[i] = fmaxf(p[i], 0.f);
}
__global__ void init_bias_k(float* __restrict__ C, const float* __restrict__ b, int n) {
  int i = blockIdx.x * blockDim.x + threadIdx.x;
  if (i < n) C[i] = b[i & (HIDC - 1)];
}

// ---------------- Laplacian edge weights ----------------
__global__ void deg_k(const int* __restrict__ src, const float* __restrict__ w,
                      float* __restrict__ deg, int E) {
  int i = blockIdx.x * blockDim.x + threadIdx.x;
  if (i < E) atomicAdd(&deg[src[i]], w[i]);
}
__global__ void lapw_k(const int* __restrict__ src, const int* __restrict__ dst,
                       const float* __restrict__ w, const float* __restrict__ deg,
                       float* __restrict__ lapw, int E) {
  int i = blockIdx.x * blockDim.x + threadIdx.x;
  if (i < E) {
    float ds = deg[src[i]], dd = deg[dst[i]];
    float rs = ds > 0.f ? rsqrtf(fmaxf(ds, 1e-30f)) : 0.f;
    float rd = dd > 0.f ? rsqrtf(fmaxf(dd, 1e-30f)) : 0.f;
    lapw[i] = -(rs * w[i] * rd);
  }
}

// ---------------- weight convert: W[kc][k][c] f32 -> Wt[kc][c][k] bf16 ----------------
__global__ void convw_k(const float* __restrict__ W, __bf16* __restrict__ Wt,
                        int F, int total) {
  int i = blockIdx.x * blockDim.x + threadIdx.x;
  if (i < total) {
    int kc = i / (F * HIDC);
    int r  = i - kc * F * HIDC;
    int k  = r / HIDC;
    int c  = r - k * HIDC;
    Wt[(size_t)kc * HIDC * F + (size_t)c * F + k] = (__bf16)W[i];
  }
}

// ---------------- edge propagation: Out[dst] += scale*lapw*Z[src] ----------------
__global__ void prop_k(const int* __restrict__ src, const int* __restrict__ dst,
                       const float* __restrict__ lapw, const float* __restrict__ Z,
                       float* __restrict__ Out, int E, int F, int lg, float scale) {
  int gid = blockIdx.x * blockDim.x + threadIdx.x;
  int tot = E << lg;                       // E * F/4
  if (gid >= tot) return;
  int e = gid >> lg;
  int q = (gid & ((1 << lg) - 1)) << 2;
  float w = scale * lapw[e];
  int s = src[e], d = dst[e];
  const float4 z = *(const float4*)(Z + (size_t)s * F + q);
  float* o = Out + (size_t)d * F + q;
  atomicAdd(o + 0, w * z.x);
  atomicAdd(o + 1, w * z.y);
  atomicAdd(o + 2, w * z.z);
  atomicAdd(o + 3, w * z.w);
}

// ---------------- WMMA GEMM: C[N x HIDC] += A[N x F] (f32->bf16) @ Wt(bf16, col-major) ----
// Block = 8 waves; each wave computes a 16x32 slab (two 16x16 WMMA tiles) so
// one block covers all 256 output columns of a 16-row stripe. Two independent
// accumulator chains per wave hide the WMMA->WMMA hazard latency.
__global__ __launch_bounds__(256)
void gemm_bf16_wmma(const float* __restrict__ A, const __bf16* __restrict__ Wt,
                    float* __restrict__ C, int N, int F) {
  const int lane = threadIdx.x & 31;
  const int wave = threadIdx.x >> 5;
  const int l15  = lane & 15;
  const int hi   = lane >> 4;
  const int m0   = blockIdx.x * 16;
  const int c0   = wave * 32 + l15;   // columns of tile 0
  const int c1   = c0 + 16;           // columns of tile 1

  // C/D layout: VGPR j holds row m0 + hi*8 + j, column = lane&15.
  v8f acc0, acc1;
#pragma unroll
  for (int j = 0; j < 8; ++j) {
    int row = m0 + hi * 8 + j; if (row >= N) row = N - 1;   // branch-free clamp
    acc0[j] = C[(size_t)row * HIDC + c0];
    acc1[j] = C[(size_t)row * HIDC + c1];
  }

  int ra = m0 + l15; if (ra >= N) ra = N - 1;
  const float*  Arow = A  + (size_t)ra * F;
  const __bf16* B0   = Wt + (size_t)c0 * F;   // transposed weights: contiguous K
  const __bf16* B1   = Wt + (size_t)c1 * F;

  for (int kb = 0; kb < F; kb += 32) {
    // A 16-bit operand layout: lanes 0-15 hold K {0..7,16..23}, lanes 16-31 {8..15,24..31}
    const float* p = Arow + kb + (hi << 3);
    if (kb + 32 < F) __builtin_prefetch(p + 32, 0, 1);      // global_prefetch_b8
    float4 a0 = *(const float4*)(p);
    float4 a1 = *(const float4*)(p + 4);
    float4 a2 = *(const float4*)(p + 16);
    float4 a3 = *(const float4*)(p + 20);
    v16bf af = { (__bf16)a0.x, (__bf16)a0.y, (__bf16)a0.z, (__bf16)a0.w,
                 (__bf16)a1.x, (__bf16)a1.y, (__bf16)a1.z, (__bf16)a1.w,
                 (__bf16)a2.x, (__bf16)a2.y, (__bf16)a2.z, (__bf16)a2.w,
                 (__bf16)a3.x, (__bf16)a3.y, (__bf16)a3.z, (__bf16)a3.w };

    // B operand: lane = column, lanes 0-15 hold K kb+{0..7,16..23}, lanes 16-31 kb+{8..15,24..31}
    const __bf16* q0 = B0 + kb + (hi << 3);
    const __bf16* q1 = B1 + kb + (hi << 3);
    v8bf b00 = *(const v8bf*)(q0);
    v8bf b01 = *(const v8bf*)(q0 + 16);
    v8bf b10 = *(const v8bf*)(q1);
    v8bf b11 = *(const v8bf*)(q1 + 16);
    v16bf bv0 = __builtin_shufflevector(b00, b01, 0,1,2,3,4,5,6,7,8,9,10,11,12,13,14,15);
    v16bf bv1 = __builtin_shufflevector(b10, b11, 0,1,2,3,4,5,6,7,8,9,10,11,12,13,14,15);

    acc0 = __builtin_amdgcn_wmma_f32_16x16x32_bf16(
        false, af, false, bv0, (short)0, acc0, false, false);
    acc1 = __builtin_amdgcn_wmma_f32_16x16x32_bf16(
        false, af, false, bv1, (short)0, acc1, false, false);
  }

#pragma unroll
  for (int j = 0; j < 8; ++j) {
    int row = m0 + hi * 8 + j; if (row >= N) row = N - 1;
    C[(size_t)row * HIDC + c0] = acc0[j];
    C[(size_t)row * HIDC + c1] = acc1[j];
  }
}

// ---------------- pooling + MLP head ----------------
__global__ void pool_sum_k(const float* __restrict__ h, const int* __restrict__ batch,
                           float* __restrict__ Gsum, int N) {
  int i = blockIdx.x * blockDim.x + threadIdx.x;
  int tot = N * (HIDC / 4);
  if (i >= tot) return;
  int node = i >> 6;                 // HIDC/4 == 64
  int q    = (i & 63) << 2;
  int b = batch[node];
  const float4 v = *(const float4*)(h + (size_t)node * HIDC + q);
  float* o = Gsum + (size_t)b * HIDC + q;
  atomicAdd(o + 0, v.x);
  atomicAdd(o + 1, v.y);
  atomicAdd(o + 2, v.z);
  atomicAdd(o + 3, v.w);
}
__global__ void pool_cnt_k(const int* __restrict__ batch, float* __restrict__ cnt, int N) {
  int i = blockIdx.x * blockDim.x + threadIdx.x;
  if (i < N) atomicAdd(&cnt[batch[i]], 1.f);
}
__global__ void pool_fin_k(float* __restrict__ Gsum, const float* __restrict__ cnt, int n) {
  int i = blockIdx.x * blockDim.x + threadIdx.x;
  if (i < n) {
    float c = fmaxf(cnt[i >> 8], 1.f);   // HIDC == 256
    Gsum[i] = fmaxf(Gsum[i] / c, 0.f);   // mean + relu
  }
}
__global__ void head1_k(const float* __restrict__ G, const float* __restrict__ lw1,
                        const float* __restrict__ lb1, float* __restrict__ Y1) {
  int g = blockIdx.x, j = threadIdx.x;
  float acc = lb1[j];
  for (int i = 0; i < HIDC; ++i) acc += G[g * HIDC + i] * lw1[i * HIDC + j];
  Y1[g * HIDC + j] = acc;
}
__global__ void head2_k(const float* __restrict__ Y1, const float* __restrict__ lw2,
                        const float* __restrict__ lb2, float* __restrict__ out) {
  __shared__ float red[HIDC];
  int g = blockIdx.x, j = threadIdx.x;
  red[j] = Y1[g * HIDC + j] * lw2[j];
  __syncthreads();
  for (int s = HIDC / 2; s > 0; s >>= 1) {
    if (j < s) red[j] += red[j + s];
    __syncthreads();
  }
  if (j == 0) {
    float v = red[0] + lb2[0];
    out[g] = 1.f / (1.f + expf(-v));
  }
}

// ---------------------------------------------------------------------------
extern "C" void kernel_launch(void* const* d_in, const int* in_sizes, int n_in,
                              void* d_out, int out_size, void* d_ws, size_t ws_size,
                              hipStream_t stream) {
  (void)n_in; (void)out_size; (void)ws_size;
  const float* x   = (const float*)d_in[0];
  const int*   ei  = (const int*)d_in[1];
  const float* ew  = (const float*)d_in[2];
  const int*   bat = (const int*)d_in[3];
  const float* W1  = (const float*)d_in[4];
  const float* b1  = (const float*)d_in[5];
  const float* W2  = (const float*)d_in[6];
  const float* b2  = (const float*)d_in[7];
  const float* W3  = (const float*)d_in[8];
  const float* b3  = (const float*)d_in[9];
  const float* lw1 = (const float*)d_in[10];
  const float* lb1 = (const float*)d_in[11];
  const float* lw2 = (const float*)d_in[12];
  const float* lb2 = (const float*)d_in[13];
  float* out = (float*)d_out;

  const int N = in_sizes[0] / FIN;
  const int E = in_sizes[2];
  const int* src = ei;
  const int* dst = ei + E;

  // ---- workspace carve ----
  char* w = (char*)d_ws;
  auto carve = [&](size_t bytes) {
    void* p = (void*)w;
    w += (bytes + 255) & ~(size_t)255;
    return p;
  };
  const size_t nodeH = (size_t)N * HIDC;
  float*  Ta   = (float*)carve(nodeH * 4);
  float*  Tb   = (float*)carve(nodeH * 4);
  float*  O1   = (float*)carve(nodeH * 4);
  float*  O2   = (float*)carve(nodeH * 4);
  float*  deg  = (float*)carve((size_t)N * 4);
  float*  lapw = (float*)carve((size_t)E * 4);
  __bf16* W1t  = (__bf16*)carve((size_t)KCHEB * FIN  * HIDC * 2);
  __bf16* W2t  = (__bf16*)carve((size_t)KCHEB * HIDC * HIDC * 2);
  __bf16* W3t  = (__bf16*)carve((size_t)KCHEB * HIDC * HIDC * 2);
  float*  Gsum = (float*)carve((size_t)NGRAPH * HIDC * 4);
  float*  Gcnt = (float*)carve((size_t)NGRAPH * 4);
  float*  Y1   = (float*)carve((size_t)NGRAPH * HIDC * 4);

  auto blocks = [](int n) { return (n + 255) / 256; };

  // ---- Laplacian weights ----
  fill_zero_k<<<blocks(N), 256, 0, stream>>>(deg, N);
  deg_k<<<blocks(E), 256, 0, stream>>>(src, ew, deg, E);
  lapw_k<<<blocks(E), 256, 0, stream>>>(src, dst, ew, deg, lapw, E);

  // ---- weights -> transposed bf16 (done once per call, tiny) ----
  convw_k<<<blocks(KCHEB * FIN  * HIDC), 256, 0, stream>>>(W1, W1t, FIN,  KCHEB * FIN  * HIDC);
  convw_k<<<blocks(KCHEB * HIDC * HIDC), 256, 0, stream>>>(W2, W2t, HIDC, KCHEB * HIDC * HIDC);
  convw_k<<<blocks(KCHEB * HIDC * HIDC), 256, 0, stream>>>(W3, W3t, HIDC, KCHEB * HIDC * HIDC);

  auto gemm = [&](const float* A, const __bf16* Wt, float* C, int F) {
    dim3 g((N + 15) / 16);   // one block covers 16 rows x all 256 columns
    gemm_bf16_wmma<<<g, 256, 0, stream>>>(A, Wt, C, N, F);
  };
  auto prop = [&](const float* Z, float* O, int F, float scale) {
    int lg = (F == 256) ? 6 : 5;   // log2(F/4)
    prop_k<<<blocks(E << lg), 256, 0, stream>>>(src, dst, lapw, Z, O, E, F, lg, scale);
  };

  // Chebyshev layer: O = sum_k Tx_k @ W[k] + b, with Tx2 = 2*prop(Tx1) - Tx0
  // folded as "init accumulator to -Tx0, scatter 2*w*z" (only 2 temp buffers).
  auto layer = [&](const float* in, int F, const __bf16* Wt, const float* bias,
                   float* O, bool relu) {
    const size_t HF = (size_t)HIDC * F;
    const int nF = N * F;
    init_bias_k<<<blocks(N * HIDC), 256, 0, stream>>>(O, bias, N * HIDC);
    gemm(in, Wt, O, F);                                           // k=0 (Tx0 = in)
    fill_zero_k<<<blocks(nF), 256, 0, stream>>>(Ta, nF);
    prop(in, Ta, F, 1.f);                                         // Tx1
    gemm(Ta, Wt + HF, O, F);
    copy_neg_k<<<blocks(nF), 256, 0, stream>>>(in, Tb, nF);
    prop(Ta, Tb, F, 2.f);                                         // Tx2
    gemm(Tb, Wt + 2 * HF, O, F);
    neg_inplace_k<<<blocks(nF), 256, 0, stream>>>(Ta, nF);
    prop(Tb, Ta, F, 2.f);                                         // Tx3
    gemm(Ta, Wt + 3 * HF, O, F);
    neg_inplace_k<<<blocks(nF), 256, 0, stream>>>(Tb, nF);
    prop(Ta, Tb, F, 2.f);                                         // Tx4
    gemm(Tb, Wt + 4 * HF, O, F);
    if (relu) relu_k<<<blocks(N * HIDC), 256, 0, stream>>>(O, N * HIDC);
  };

  layer(x,  FIN,  W1t, b1, O1, true);
  layer(O1, HIDC, W2t, b2, O2, false);
  layer(O2, HIDC, W3t, b3, O1, true);

  // ---- global mean pool + head ----
  fill_zero_k<<<blocks(NGRAPH * HIDC), 256, 0, stream>>>(Gsum, NGRAPH * HIDC);
  fill_zero_k<<<1, 256, 0, stream>>>(Gcnt, NGRAPH);
  pool_sum_k<<<blocks(N * (HIDC / 4)), 256, 0, stream>>>(O1, bat, Gsum, N);
  pool_cnt_k<<<blocks(N), 256, 0, stream>>>(bat, Gcnt, N);
  pool_fin_k<<<blocks(NGRAPH * HIDC), 256, 0, stream>>>(Gsum, Gcnt, NGRAPH * HIDC);
  head1_k<<<NGRAPH, HIDC, 0, stream>>>(Gsum, lw1, lb1, Y1);
  head2_k<<<NGRAPH, HIDC, 0, stream>>>(Y1, lw2, lb2, out);
}
